// VQDic_7825430413747
// MI455X (gfx1250) — matrix operand order repaired and checked by reading, not compiled, not run on hardware
//
#include <hip/hip_runtime.h>
#include <hip/hip_bf16.h>

typedef __attribute__((ext_vector_type(16))) _Float16 v16h;
typedef __attribute__((ext_vector_type(8)))  _Float16 v8h;
typedef __attribute__((ext_vector_type(8)))  float    v8f;
typedef __attribute__((ext_vector_type(4)))  unsigned u32x4;
typedef __attribute__((ext_vector_type(8)))  int      i32x8;
typedef __attribute__((ext_vector_type(4)))  int      i32x4;

#define Bn   32
#define Fn   256
#define HWn  1024
#define Kn   1024
#define Mn   (Bn * HWn)          // 32768 rows
#define ROW_HALF  264            // padded LDS row stride in halves (528 B)
#define TILE_HALF (16 * ROW_HALF)

// ---------------------------------------------------------------------------
// TDM: copy one B tile (16 codewords x 512 bytes) global->LDS, inserting
// 16 B of padding after every 512 B row (pad_interval=6 -> 128 DWORDs,
// pad_amount=3 -> 4 DWORDs) so LDS row stride is 528 B (bank-conflict free).
// ---------------------------------------------------------------------------
__device__ inline void tdm_load_tile(const void* gsrc, unsigned lds_byte_off) {
  const unsigned long long ga = (unsigned long long)(size_t)gsrc;
  u32x4 g0;
  g0[0] = 1u;                                            // count=1, user mode
  g0[1] = lds_byte_off;                                  // lds_addr (bytes)
  g0[2] = (unsigned)ga;                                  // global_addr[31:0]
  g0[3] = (unsigned)((ga >> 32) & 0x01FFFFFFu) | (2u << 30); // addr[56:32] | type=2
  i32x8 g1;
  // data_size=2 (4B units) | pad_enable | pad_interval=6 | pad_amount=3
  g1[0] = (int)((2u << 16) | (1u << 20) | (6u << 22) | (3u << 25));
  const unsigned td0 = 128u, td1 = 16u;        // tensor dims (dwords, rows)
  const unsigned tile0 = 128u, tile1 = 16u;    // tile dims
  const unsigned long long s0 = 128ull;        // tensor_dim0_stride (dwords)
  const unsigned long long s1 = 2048ull;       // tensor_dim1_stride (unused, 2D)
  g1[1] = (int)((td0 & 0xFFFFu) << 16);                       // [15:0]=atomic addr
  g1[2] = (int)(((td0 >> 16) & 0xFFFFu) | ((td1 & 0xFFFFu) << 16));
  g1[3] = (int)(((td1 >> 16) & 0xFFFFu) | (tile0 << 16));
  g1[4] = (int)(tile1);                                       // tile_dim2 = 0
  g1[5] = (int)(s0 & 0xFFFFFFFFull);
  g1[6] = (int)(((s0 >> 32) & 0xFFFFull) | ((s1 & 0xFFFFull) << 16));
  g1[7] = (int)((s1 >> 16) & 0xFFFFFFFFull);
  i32x4 g2 = {};
  i32x4 g3 = {};
  i32x8 g4 = {};
  __builtin_amdgcn_tensor_load_to_lds(g0, g1, g2, g3, g4, 0);
}

// ---------------------------------------------------------------------------
// Prep: z_dic (F,K) f32 -> Bws (K,F) f16  (B operand, k-major, f contiguous)
// ---------------------------------------------------------------------------
__global__ __launch_bounds__(256) void vq_prep_b(const float* __restrict__ zd,
                                                 _Float16* __restrict__ Bws) {
  const int t = blockIdx.x * 256 + threadIdx.x;   // t = k*256 + f
  const int f = t & (Fn - 1);
  const int k = t >> 8;
  Bws[t] = (_Float16)zd[(size_t)f * Kn + k];
}

// d2[k] = sum_f z_dic[f,k]^2  (full f32 precision)
__global__ __launch_bounds__(256) void vq_prep_d2(const float* __restrict__ zd,
                                                  float* __restrict__ d2) {
  const int k = blockIdx.x * 256 + threadIdx.x;
  float s = 0.f;
  for (int f = 0; f < Fn; ++f) {
    float v = zd[(size_t)f * Kn + k];
    s = fmaf(v, v, s);
  }
  d2[k] = s;
}

// ---------------------------------------------------------------------------
// Fused GEMM (zd = z . z_dic) + argmin_k (d2[k] - 2*zd)
// 8 waves/block, each wave owns a 16-row M tile, loops over all 64 N tiles.
// ---------------------------------------------------------------------------
__global__ __launch_bounds__(256) void vq_gemm(const float* __restrict__ z,
                                               const _Float16* __restrict__ Bws,
                                               const float* __restrict__ d2,
                                               int* __restrict__ idx_out) {
  __shared__ __align__(16) _Float16 smemB[2 * TILE_HALF];
  __shared__ float smemD2[Kn];

  const int tid  = threadIdx.x;
  const int wave = tid >> 5;
  const int lane = tid & 31;
  const int lh   = lane & 15;
  const bool hi  = lane >= 16;

  // preload d2 into LDS (coalesced)
  #pragma unroll
  for (int i = 0; i < Kn / 256; ++i)
    smemD2[tid + i * 256] = d2[tid + i * 256];

  // Load + convert this wave's A tile (16 m-rows x 256 f) into the exact
  // 16-bit 16x32 A-matrix lane layout (8 x v16h = 64 VGPRs, reused 64x).
  const int mbase = blockIdx.x * 128 + wave * 16;
  const int m     = mbase + lh;
  const float* zrow = z + (size_t)(m >> 10) * (Fn * HWn) + (m & (HWn - 1));
  v16h a[8];
  #pragma unroll
  for (int kc = 0; kc < 8; ++kc) {
    const int fb = kc * 32 + (hi ? 8 : 0);
    #pragma unroll
    for (int e = 0; e < 8; ++e) {
      a[kc][e]     = (_Float16)zrow[(fb + e) * HWn];        // K = 0..7 / 8..15
      a[kc][e + 8] = (_Float16)zrow[(fb + 16 + e) * HWn];   // K = 16..23 / 24..31
    }
  }

  // first B tile via TDM
  if (wave == 0)
    tdm_load_tile(Bws, (unsigned)(size_t)&smemB[0]);

  float best[8];
  int   bestk[8];
  #pragma unroll
  for (int r = 0; r < 8; ++r) { best[r] = 3.4e38f; bestk[r] = 0; }

  for (int nt = 0; nt < 64; ++nt) {
    __builtin_amdgcn_s_wait_tensorcnt(0);  // no-op for waves with cnt==0
    __syncthreads();                       // tile nt visible to all waves
    if (wave == 0 && nt + 1 < 64)
      tdm_load_tile(Bws + (size_t)(nt + 1) * 16 * Fn,
                    (unsigned)(size_t)&smemB[((nt + 1) & 1) * TILE_HALF]);

    // B fragment: lane<16 holds f=kc*32+{0..7,16..23} of column lh, lane>=16 holds +8
    const _Float16* bp =
        &smemB[(nt & 1) * TILE_HALF + lh * ROW_HALF + (hi ? 8 : 0)];
    v8f acc = {};
    #pragma unroll
    for (int kc = 0; kc < 8; ++kc) {
      v8h blo = *(const v8h*)(bp + kc * 32);
      v8h bhi = *(const v8h*)(bp + kc * 32 + 16);
      v16h bf = __builtin_shufflevector(blo, bhi, 0, 1, 2, 3, 4, 5, 6, 7,
                                        8, 9, 10, 11, 12, 13, 14, 15);
      acc = __builtin_amdgcn_wmma_f32_16x16x32_f16(
          false, a[kc], false, bf, (short)0, acc, false, false);
    }

    // score = d2[k] - 2*zd ; lane handles column k, rows mbase+r(+8)
    const int   kk  = nt * 16 + lh;
    const float d2k = smemD2[kk];
    #pragma unroll
    for (int r = 0; r < 8; ++r) {
      float s = __builtin_fmaf(-2.0f, acc[r], d2k);
      if (s < best[r]) { best[r] = s; bestk[r] = kk; }
    }
  }

  // reduce (score,k) across the 16 lanes of each half; tie -> smallest k
  #pragma unroll
  for (int r = 0; r < 8; ++r) {
    float s  = best[r];
    int   kk = bestk[r];
    #pragma unroll
    for (int off = 8; off >= 1; off >>= 1) {
      float so = __shfl_xor(s, off, 32);
      int   ko = __shfl_xor(kk, off, 32);
      if (so < s || (so == s && ko < kk)) { s = so; kk = ko; }
    }
    if (lh == 0)
      idx_out[mbase + (hi ? 8 : 0) + r] = kk;
  }
}

// ---------------------------------------------------------------------------
// Gather: quantized[b,f,hw] = z_dic[f, idx[b,hw]]  (full f32 dictionary)
// ---------------------------------------------------------------------------
__global__ __launch_bounds__(256) void vq_gather(const float* __restrict__ zd,
                                                 const int* __restrict__ idx,
                                                 float* __restrict__ out) {
  const int t  = blockIdx.x * 256 + threadIdx.x;
  const int hw = t & (HWn - 1);
  const int f  = (t >> 10) & (Fn - 1);
  const int b  = t >> 18;
  const int m  = (b << 10) | hw;
  out[t] = zd[(size_t)f * Kn + idx[m]];
}

extern "C" void kernel_launch(void* const* d_in, const int* in_sizes, int n_in,
                              void* d_out, int out_size, void* d_ws,
                              size_t ws_size, hipStream_t stream) {
  (void)in_sizes; (void)n_in; (void)out_size; (void)ws_size;
  const float* z  = (const float*)d_in[0];   // (32,256,32,32) f32
  const float* zd = (const float*)d_in[1];   // (256,1024) f32

  float* quant = (float*)d_out;                         // 8,388,608 f32
  int*   idx   = (int*)(quant + (size_t)Bn * Fn * HWn); // 32,768 int32

  _Float16* Bws = (_Float16*)d_ws;                      // 512 KB f16 (K,F)
  float*    d2  = (float*)((char*)d_ws + (size_t)Kn * Fn * sizeof(_Float16));

  vq_prep_b <<<(Kn * Fn) / 256, 256, 0, stream>>>(zd, Bws);
  vq_prep_d2<<<Kn / 256,       256, 0, stream>>>(zd, d2);
  vq_gemm   <<<Mn / 128,       256, 0, stream>>>(z, Bws, d2, idx);
  vq_gather <<<(Bn * Fn * HWn) / 256, 256, 0, stream>>>(zd, idx, quant);
}
